// NeuralTuringMachine_46162308497797
// MI455X (gfx1250) — matrix operand order repaired and checked by reading, not compiled, not run on hardware
//
#include <hip/hip_runtime.h>
#include <hip/hip_bf16.h>

// ---------------------------------------------------------------------------
// NTM forward for MI455X (gfx1250, wave32, WMMA bf16 f32-accumulate).
//
// Pipeline (all on `stream`):
//   1. pack kernels: swizzle W_ih/W_hh/W_read/W_out/x into WMMA lane layouts
//      (ISA 7.12.2), fp32 -> bf16 (RNE).
//   2. xproj_k  : Xproj[t] = x_t @ W_ih^T + b_ih  (parallel over all t, WMMA)
//   3. scan_k   : persistent 32-WG kernel; per step one fused
//                 gates-WMMA + LSTM stage, c-state held in VGPRs, one grid
//                 barrier per step. Writes h_t in packed-A bf16 layout.
//   4. rp_k     : rp = H @ W_read^T (batched WMMA over all t)
//   5. attn_k   : cosine attention + softmax (fp32 VALU), writes read_vec
//                 into the packed-A buffer (K-slots 1024..1151)
//   6. out_k    : out = [h, read_vec] @ W_out^T (batched WMMA), + b_out
// ---------------------------------------------------------------------------

typedef __attribute__((ext_vector_type(16))) __bf16 v16bf;
typedef __attribute__((ext_vector_type(8)))  float  v8f;

#define NB   16      // batch
#define NS   512     // seq len
#define NI   512     // input dim
#define NH   1024    // hidden
#define NG   4096    // 4*NH
#define ND   128     // memory width
#define NMEM 1024    // memory rows
#define NO   512     // output dim
#define NKOUT 1152   // NH + ND

// ---- workspace layout (bytes) ---------------------------------------------
static constexpr size_t OFF_CNT   = 0;                                   // grid-sync counter
static constexpr size_t OFF_WHH   = 256;                                 // 256nt x 32kc x 512 bf16
static constexpr size_t OFF_WIH   = OFF_WHH  + (size_t)256*32*512*2;     // 256nt x 16kc x 512
static constexpr size_t OFF_WRD   = OFF_WIH  + (size_t)256*16*512*2;     // 9nt x 32kc x 512 (zero-padded)
static constexpr size_t OFF_WOUT  = OFF_WRD  + (size_t)9*32*512*2;       // 32nt x 36kc x 512
static constexpr size_t OFF_XA    = OFF_WOUT + (size_t)32*36*512*2;      // 512t x 16kc x 512
static constexpr size_t OFF_XPROJ = OFF_XA   + (size_t)512*16*512*2;     // 512t x 256nt x 256 f32 (C layout)
static constexpr size_t OFF_HPA   = OFF_XPROJ+ (size_t)512*256*256*4;    // 512t x 36kc x 512 bf16 (packed A)
static constexpr size_t OFF_RP    = OFF_HPA  + (size_t)512*36*512*2;     // 8192 x 144 f32
static constexpr size_t OFF_MNORM = OFF_RP   + (size_t)8192*144*4;       // 1024 f32
// total ~181 MB; assumed <= ws_size.

__device__ __forceinline__ unsigned short f2bf(float x) {
  unsigned u = __float_as_uint(x);
  unsigned r = u + 0x7FFFu + ((u >> 16) & 1u);   // round-to-nearest-even
  return (unsigned short)(r >> 16);
}
__device__ __forceinline__ float sigm_(float x) { return 1.0f / (1.0f + __expf(-x)); }
__device__ __forceinline__ float tanh_(float x) {
  x = fminf(fmaxf(x, -15.0f), 15.0f);
  float e = __expf(2.0f * x);
  return (e - 1.0f) / (e + 1.0f);
}
__device__ __forceinline__ v16bf ldbf(const unsigned short* p) {
  return *reinterpret_cast<const v16bf*>(p);
}
#define WMMA_BF16(A, Bm, C) \
  __builtin_amdgcn_wmma_f32_16x16x32_bf16(false, (A), false, (Bm), (short)0, (C), false, false)

// ---- pack B operand: dst[(nt*kcs+kc)*512 + lane*16 + e] = W[n,k] ----------
// B layout (32x16 KxN, 16-bit): lane = n%16 + 16*h1, k%32 = (e&7) + 8*h1 + 16*(e>>3).
__global__ void pack_b_k(const float* __restrict__ W, unsigned short* __restrict__ dst,
                         int R, int C, int kcs) {
  int tile = blockIdx.x;
  int nt = tile / kcs, kc = tile % kcs;
  for (int s = 0; s < 2; ++s) {
    int idx  = threadIdx.x + s * 256;          // 0..511
    int lane = idx >> 4, e = idx & 15;
    int n = nt * 16 + (lane & 15);
    int k = kc * 32 + (e & 7) + 8 * (lane >> 4) + 16 * (e >> 3);
    float v = (n < R && k < C) ? W[(size_t)n * C + k] : 0.0f;
    dst[(size_t)tile * 512 + idx] = f2bf(v);
  }
}

// ---- pack A operand from x: A rows = batch, cols = input k -----------------
// A layout (16x32 MxK, 16-bit): lane = m + 16*h1, kk = (e&7) + 8*h1 + 16*(e>>3)
__global__ void pack_x_k(const float* __restrict__ x, unsigned short* __restrict__ dst) {
  int tile = blockIdx.x;                       // t*16 + kc
  int t = tile >> 4, kc = tile & 15;
  for (int s = 0; s < 2; ++s) {
    int idx  = threadIdx.x + s * 256;
    int lane = idx >> 4, e = idx & 15;
    int m  = lane & 15;
    int kk = (e & 7) + 8 * (lane >> 4) + 16 * (e >> 3);
    int k  = kc * 32 + kk;
    dst[(size_t)tile * 512 + idx] = f2bf(x[((size_t)m * NS + t) * NI + k]);
  }
}

__global__ void mem_norm_k(const float* __restrict__ mem, float* __restrict__ mn) {
  int j = blockIdx.x * blockDim.x + threadIdx.x;
  if (j < NMEM) {
    float s = 0.0f;
    for (int d = 0; d < ND; ++d) { float v = mem[(size_t)j * ND + d]; s += v * v; }
    mn[j] = fmaxf(sqrtf(s), 1e-8f);
  }
}

// ---- Xproj[t] = x_t @ W_ih^T + b_ih, stored in WMMA C layout ---------------
__global__ void xproj_k(const unsigned short* __restrict__ xa,
                        const unsigned short* __restrict__ wih,
                        const float* __restrict__ b_ih, float* __restrict__ xprojp) {
  int gw   = (blockIdx.x * blockDim.x + threadIdx.x) >> 5;  // global wave
  int lane = threadIdx.x & 31;
  int t = gw >> 8, nt = gw & 255;
  const v16bf* A  = (const v16bf*)(xa  + (size_t)t  * 16 * 512);
  const v16bf* Bm = (const v16bf*)(wih + (size_t)nt * 16 * 512);
  v8f acc = {0.f,0.f,0.f,0.f,0.f,0.f,0.f,0.f};
  for (int kc = 0; kc < 16; ++kc)
    acc = WMMA_BF16(A[kc * 32 + lane], Bm[kc * 32 + lane], acc);
  float bias = b_ih[nt * 16 + (lane & 15)];
  float* out = xprojp + ((size_t)t * 256 + nt) * 256 + lane * 8;
  for (int q = 0; q < 8; ++q) out[q] = acc[q] + bias;
}

// ---- grid-wide barrier (persistent kernel) ---------------------------------
__device__ __forceinline__ void grid_sync(unsigned* cnt, unsigned target) {
  __syncthreads();
  if (threadIdx.x == 0) {
    __threadfence();                          // release: make h_t visible
    atomicAdd(cnt, 1u);
    while (__hip_atomic_load(cnt, __ATOMIC_RELAXED, __HIP_MEMORY_SCOPE_AGENT) < target)
      __builtin_amdgcn_s_sleep(1);
    __threadfence();                          // acquire
  }
  __syncthreads();
}

// ---- persistent LSTM scan: 32 WGs x 256 thr, c-state in VGPRs --------------
// wave w: jt = wg*2 + (w>>2) (hidden tile 0..63), ks = w&3 (K quarter).
// Each wave does 4 gate tiles (i,f,g,o) sharing A; LDS reduce; owner (ks==0)
// applies LSTM and writes h_t (packed-A bf16). One grid barrier per step.
__global__ void scan_k(const unsigned short* __restrict__ whh,
                       const float* __restrict__ xprojp,
                       const float* __restrict__ b_hh,
                       unsigned short* __restrict__ hpa,
                       unsigned* cnt) {
  __shared__ float part[8][4][32][8];                 // 32 KB
  const int tid = threadIdx.x, lane = tid & 31, w = tid >> 5;
  const int wg = blockIdx.x;                          // 0..31
  const int jt = wg * 2 + (w >> 2);                   // 0..63
  const int ks = w & 3;
  const int nl = lane & 15;
  // h-scatter mapping (constant per lane): column k = jt*16 + nl
  const int kk  = ((jt & 1) << 4) | nl;
  const int h1  = (kk >> 3) & 1;
  const int eo  = (kk & 7) + ((kk >> 4) << 3);
  const int kcw = jt >> 1;
  const size_t b0 = ((size_t)(0 * 64 + jt) * 32) * 512;
  const size_t b1 = ((size_t)(1 * 64 + jt) * 32) * 512;
  const size_t b2 = ((size_t)(2 * 64 + jt) * 32) * 512;
  const size_t b3 = ((size_t)(3 * 64 + jt) * 32) * 512;

  v8f c = {0.f,0.f,0.f,0.f,0.f,0.f,0.f,0.f};          // cell state, lives here 512 steps

  for (int t = 0; t < NS; ++t) {
    v8f d0 = {0.f,0.f,0.f,0.f,0.f,0.f,0.f,0.f};
    v8f d1 = d0, d2 = d0, d3 = d0;
    if (t > 0) {
      const v16bf* A = (const v16bf*)(hpa + (size_t)(t - 1) * 36 * 512);
      for (int kci = 0; kci < 8; ++kci) {
        int kc = ks * 8 + kci;
        v16bf a = A[kc * 32 + lane];
        __builtin_prefetch(whh + b0 + (size_t)(kc + 1) * 512 + lane * 16, 0, 1);
        v16bf q0 = ldbf(whh + b0 + (size_t)kc * 512 + lane * 16);
        v16bf q1 = ldbf(whh + b1 + (size_t)kc * 512 + lane * 16);
        v16bf q2 = ldbf(whh + b2 + (size_t)kc * 512 + lane * 16);
        v16bf q3 = ldbf(whh + b3 + (size_t)kc * 512 + lane * 16);
        d0 = WMMA_BF16(a, q0, d0);
        d1 = WMMA_BF16(a, q1, d1);
        d2 = WMMA_BF16(a, q2, d2);
        d3 = WMMA_BF16(a, q3, d3);
      }
    }
    for (int q = 0; q < 8; ++q) {
      part[w][0][lane][q] = d0[q];
      part[w][1][lane][q] = d1[q];
      part[w][2][lane][q] = d2[q];
      part[w][3][lane][q] = d3[q];
    }
    __syncthreads();

    if (ks == 0) {
      v8f gate[4];
      for (int g4 = 0; g4 < 4; ++g4) {
        int ntile = g4 * 64 + jt;
        float bias = b_hh[ntile * 16 + nl];
        const float* xp = xprojp + ((size_t)t * 256 + ntile) * 256 + lane * 8;
        for (int q = 0; q < 8; ++q)
          gate[g4][q] = xp[q] + bias
                      + part[w + 0][g4][lane][q] + part[w + 1][g4][lane][q]
                      + part[w + 2][g4][lane][q] + part[w + 3][g4][lane][q];
      }
      // LSTM pointwise on the 16x16 tile; write h_t in packed-A bf16 layout
      unsigned short* dst = hpa + ((size_t)t * 36 + kcw) * 512 + eo;
      for (int q = 0; q < 8; ++q) {
        float iv = sigm_(gate[0][q]), fv = sigm_(gate[1][q]);
        float gv = tanh_(gate[2][q]), ov = sigm_(gate[3][q]);
        float cc = fv * c[q] + iv * gv;
        c[q] = cc;
        float hv = ov * tanh_(cc);
        int m = q + ((lane >> 4) << 3);
        int lane_dst = m + (h1 << 4);
        dst[lane_dst * 16] = f2bf(hv);
      }
    }
    grid_sync(cnt, (unsigned)(t + 1) * 32u);
  }
}

// ---- rp = H @ W_read^T + b_read (batched over t, WMMA) ---------------------
__global__ void rp_k(const unsigned short* __restrict__ hpa,
                     const unsigned short* __restrict__ wrd,
                     const float* __restrict__ b_read, float* __restrict__ rp) {
  int gw   = (blockIdx.x * blockDim.x + threadIdx.x) >> 5;
  int lane = threadIdx.x & 31;
  int t = gw / 9, nt = gw % 9;
  const v16bf* A  = (const v16bf*)(hpa + (size_t)t  * 36 * 512);
  const v16bf* Bm = (const v16bf*)(wrd + (size_t)nt * 32 * 512);
  v8f acc = {0.f,0.f,0.f,0.f,0.f,0.f,0.f,0.f};
  for (int kc = 0; kc < 32; ++kc)
    acc = WMMA_BF16(A[kc * 32 + lane], Bm[kc * 32 + lane], acc);
  int n = nt * 16 + (lane & 15);
  float bias = (n < 131) ? b_read[n] : 0.0f;
  for (int q = 0; q < 8; ++q) {
    int m = q + ((lane >> 4) << 3);
    rp[((size_t)t * 16 + m) * 144 + n] = acc[q] + bias;
  }
}

// ---- cosine attention + softmax + read_vec (fp32 VALU), per (t,m) row ------
__global__ void attn_k(const float* __restrict__ rp, const float* __restrict__ mem,
                       const float* __restrict__ mnorm, unsigned short* __restrict__ hpa) {
  __shared__ float skey[128];
  __shared__ float red[256];
  __shared__ float ebuf[1024];
  __shared__ float sscal;
  const int t = blockIdx.x >> 4, m = blockIdx.x & 15, tid = threadIdx.x;
  const float* row = rp + ((size_t)t * 16 + m) * 144;

  float kv = (tid < 128) ? row[tid] : 0.0f;
  if (tid < 128) skey[tid] = kv;
  red[tid] = kv * kv;
  __syncthreads();
  for (int s = 128; s > 0; s >>= 1) { if (tid < s) red[tid] += red[tid + s]; __syncthreads(); }
  if (tid == 0) {
    float kn = fmaxf(sqrtf(red[0]), 1e-8f);
    float sr = row[128];
    float st = (sr > 20.0f) ? sr : __logf(1.0f + __expf(sr));   // softplus
    sscal = st / kn;
  }
  __syncthreads();
  float sk = sscal;

  float z[4], zmax = -3.4e38f;
  for (int r = 0; r < 4; ++r) {
    int j = tid + r * 256;
    const float* mr = mem + (size_t)j * ND;
    float dot = 0.0f;
    for (int d = 0; d < ND; ++d) dot += skey[d] * mr[d];
    z[r] = sk * dot / mnorm[j];
    zmax = fmaxf(zmax, z[r]);
  }
  __syncthreads();
  red[tid] = zmax; __syncthreads();
  for (int s = 128; s > 0; s >>= 1) { if (tid < s) red[tid] = fmaxf(red[tid], red[tid + s]); __syncthreads(); }
  zmax = red[0]; __syncthreads();

  float ls = 0.0f;
  for (int r = 0; r < 4; ++r) {
    float ev = __expf(z[r] - zmax);
    ebuf[tid + r * 256] = ev;
    ls += ev;
  }
  red[tid] = ls; __syncthreads();
  for (int s = 128; s > 0; s >>= 1) { if (tid < s) red[tid] += red[tid + s]; __syncthreads(); }
  float sum = red[0]; __syncthreads();

  // read_vec: thread -> (d, half of j range)
  int d = tid & 127, jh = tid >> 7;
  float acc = 0.0f;
  for (int j = jh * 512; j < jh * 512 + 512; ++j)
    acc += ebuf[j] * mem[(size_t)j * ND + d];
  red[tid] = acc; __syncthreads();
  if (tid < 128) {
    float rv = (red[tid] + red[tid + 128]) / sum;
    int kkd = d & 31, kc = 32 + (d >> 5);            // k = 1024 + d
    int h1d = (kkd >> 3) & 1;
    int ed  = (kkd & 7) + ((kkd >> 4) << 3);
    int lane_dst = m + (h1d << 4);
    hpa[((size_t)t * 36 + kc) * 512 + lane_dst * 16 + ed] = f2bf(rv);
  }
}

// ---- out = [h, read_vec] @ W_out^T + b_out (batched over t, WMMA) ----------
__global__ void out_k(const unsigned short* __restrict__ hpa,
                      const unsigned short* __restrict__ wout,
                      const float* __restrict__ b_out, float* __restrict__ out) {
  int gw   = (blockIdx.x * blockDim.x + threadIdx.x) >> 5;
  int lane = threadIdx.x & 31;
  int t = gw >> 5, nt = gw & 31;
  const v16bf* A  = (const v16bf*)(hpa  + (size_t)t  * 36 * 512);
  const v16bf* Bm = (const v16bf*)(wout + (size_t)nt * 36 * 512);
  v8f acc = {0.f,0.f,0.f,0.f,0.f,0.f,0.f,0.f};
  for (int kc = 0; kc < 36; ++kc)
    acc = WMMA_BF16(A[kc * 32 + lane], Bm[kc * 32 + lane], acc);
  int n = nt * 16 + (lane & 15);
  float bias = b_out[n];
  for (int q = 0; q < 8; ++q) {
    int m = q + ((lane >> 4) << 3);
    out[((size_t)m * NS + t) * NO + n] = acc[q] + bias;
  }
}

// ---------------------------------------------------------------------------
extern "C" void kernel_launch(void* const* d_in, const int* in_sizes, int n_in,
                              void* d_out, int out_size, void* d_ws, size_t ws_size,
                              hipStream_t stream) {
  (void)in_sizes; (void)n_in; (void)out_size; (void)ws_size;
  const float* x      = (const float*)d_in[0];
  const float* memory = (const float*)d_in[1];
  const float* W_ih   = (const float*)d_in[2];
  const float* W_hh   = (const float*)d_in[3];
  const float* b_ih   = (const float*)d_in[4];
  const float* b_hh   = (const float*)d_in[5];
  const float* W_read = (const float*)d_in[6];
  const float* b_read = (const float*)d_in[7];
  const float* W_out  = (const float*)d_in[8];
  const float* b_out  = (const float*)d_in[9];
  float* out = (float*)d_out;

  char* ws = (char*)d_ws;
  unsigned*       cnt    = (unsigned*)(ws + OFF_CNT);
  unsigned short* whhP   = (unsigned short*)(ws + OFF_WHH);
  unsigned short* wihP   = (unsigned short*)(ws + OFF_WIH);
  unsigned short* wrdP   = (unsigned short*)(ws + OFF_WRD);
  unsigned short* woutP  = (unsigned short*)(ws + OFF_WOUT);
  unsigned short* xaP    = (unsigned short*)(ws + OFF_XA);
  float*          xprojP = (float*)(ws + OFF_XPROJ);
  unsigned short* hpaP   = (unsigned short*)(ws + OFF_HPA);
  float*          rpB    = (float*)(ws + OFF_RP);
  float*          mnorm  = (float*)(ws + OFF_MNORM);

  hipMemsetAsync(ws + OFF_CNT, 0, 256, stream);       // reset grid-sync counter

  pack_b_k<<<256 * 32, 256, 0, stream>>>(W_hh,   whhP,  NG,  NH,    32);
  pack_b_k<<<256 * 16, 256, 0, stream>>>(W_ih,   wihP,  NG,  NI,    16);
  pack_b_k<<<9   * 32, 256, 0, stream>>>(W_read, wrdP,  131, NH,    32);
  pack_b_k<<<32  * 36, 256, 0, stream>>>(W_out,  woutP, NO,  NKOUT, 36);
  pack_x_k<<<512 * 16, 256, 0, stream>>>(x, xaP);
  mem_norm_k<<<4, 256, 0, stream>>>(memory, mnorm);

  xproj_k<<<16384, 256, 0, stream>>>(xaP, wihP, b_ih, xprojP);     // 131072 wave-tiles
  scan_k<<<32, 256, 0, stream>>>(whhP, xprojP, b_hh, hpaP, cnt);   // persistent scan
  rp_k<<<576, 256, 0, stream>>>(hpaP, wrdP, b_read, rpB);          // 4608 wave-tiles
  attn_k<<<NS * NB, 256, 0, stream>>>(rpB, memory, mnorm, hpaP);   // 8192 rows
  out_k<<<2048, 256, 0, stream>>>(hpaP, woutP, b_out, out);        // 16384 wave-tiles
}
